// MultiHeadAttention_61967788147213
// MI455X (gfx1250) — matrix operand order repaired and checked by reading, not compiled, not run on hardware
//
#include <hip/hip_runtime.h>
#include <hip/hip_bf16.h>

#define D_IN   1024
#define D_OUT  1024
#define NHEADS 16
#define HDIM   64
#define SEQ    2048
#define BATCH  2
#define BN     (BATCH * SEQ)   // 4096 rows

typedef __attribute__((ext_vector_type(16))) __bf16 v16bf;
typedef __attribute__((ext_vector_type(8)))  __bf16 v8bf;
typedef __attribute__((ext_vector_type(8)))  float  v8f;

// ---------------------------------------------------------------------------
// WMMA helper: D = A(16x32 bf16) * B(32x16 bf16) + C(16x16 f32)
// ---------------------------------------------------------------------------
__device__ __forceinline__ v8f wmma_bf16(v16bf a, v16bf b, v8f c) {
  return __builtin_amdgcn_wmma_f32_16x16x32_bf16(
      /*neg_a=*/false, a, /*neg_b=*/false, b,
      /*c_mod=*/(short)0, c, /*reuse_a=*/false, /*reuse_b=*/false);
}

// A-matrix fragment (16x32, 16-bit): lane l holds row m0+(l&15);
// elements 0..7 = K (l>>4)*8 + 0..7 ; elements 8..15 = K (l>>4)*8 + 16..23.
// Source is row-major [m][k] with contiguous k -> two 16-byte loads.
__device__ __forceinline__ v16bf load_a_frag(const __bf16* base, int stride,
                                             int m0, int k0, int lane) {
  const __bf16* p =
      base + (size_t)(m0 + (lane & 15)) * stride + k0 + ((lane >> 4) * 8);
  v8bf lo = *(const v8bf*)p;
  v8bf hi = *(const v8bf*)(p + 16);
  v16bf r;
#pragma unroll
  for (int i = 0; i < 8; ++i) { r[i] = lo[i]; r[i + 8] = hi[i]; }
  return r;
}

// B-matrix fragment (32x16, 16-bit): lane l holds column n0+(l&15);
// elements 0..7 = K (l>>4)*8 + 0..7 ; elements 8..15 = K (l>>4)*8 + 16..23.
// Source stored "transposed" row-major [n][k] with contiguous k -> two 16-byte loads.
__device__ __forceinline__ v16bf load_b_frag(const __bf16* base, int stride,
                                             int n0, int k0, int lane) {
  const __bf16* p =
      base + (size_t)(n0 + (lane & 15)) * stride + k0 + ((lane >> 4) * 8);
  v8bf lo = *(const v8bf*)p;
  v8bf hi = *(const v8bf*)(p + 16);
  v16bf r;
#pragma unroll
  for (int i = 0; i < 8; ++i) { r[i] = lo[i]; r[i + 8] = hi[i]; }
  return r;
}

// ---------------------------------------------------------------------------
// 1) fp32 -> bf16 conversion of activations
// ---------------------------------------------------------------------------
__global__ void cvt_x_kernel(const float* __restrict__ x,
                             __bf16* __restrict__ xb, int n) {
  int i = blockIdx.x * blockDim.x + threadIdx.x;
  if (i < n) xb[i] = (__bf16)x[i];
}

// 2) weight transpose + convert: Wt[n][k] = (bf16) W[k][n], 1024x1024
__global__ void transpose_w_kernel(const float* __restrict__ W,
                                   __bf16* __restrict__ Wt) {
  int i = blockIdx.x * blockDim.x + threadIdx.x;  // i = n*1024 + k
  int n = i >> 10, k = i & 1023;
  Wt[i] = (__bf16)W[(size_t)k * D_OUT + n];
}

// ---------------------------------------------------------------------------
// 3) QKV projection: C = Xbf(4096x1024) * W^T.  Each wave owns a 32x64 tile
//    (2x4 WMMAs per 32-k step -> 8 WMMAs per 6 fragment loads, ~112 VGPRs
//    live: fits the 256-VGPR budget with no spills).
//    4 waves per block, blockIdx.z selects Q/K/V.
//    Q,K stored per-head [b][h][seq][64]; V stored transposed [b][h][64][seq].
// ---------------------------------------------------------------------------
__global__ void __launch_bounds__(128, 1)
qkv_proj_kernel(const __bf16* __restrict__ xb,
                const __bf16* __restrict__ WqT,
                const __bf16* __restrict__ WkT,
                const __bf16* __restrict__ WvT,
                __bf16* __restrict__ Q,
                __bf16* __restrict__ K,
                __bf16* __restrict__ Vt) {
  const int lane = threadIdx.x & 31;
  const int wave = threadIdx.x >> 5;
  const int z = blockIdx.z;
  const int m0 = (blockIdx.x * 4 + wave) * 32;
  const int n0 = blockIdx.y * 64;
  const __bf16* Wt = (z == 0) ? WqT : (z == 1) ? WkT : WvT;

  v8f acc[2][4] = {};
  for (int k = 0; k < D_IN; k += 32) {
    v16bf a[2], b[4];
#pragma unroll
    for (int i = 0; i < 2; ++i) a[i] = load_a_frag(xb, D_IN, m0 + i * 16, k, lane);
#pragma unroll
    for (int j = 0; j < 4; ++j) b[j] = load_b_frag(Wt, D_IN, n0 + j * 16, k, lane);
#pragma unroll
    for (int i = 0; i < 2; ++i)
#pragma unroll
      for (int j = 0; j < 4; ++j) acc[i][j] = wmma_bf16(a[i], b[j], acc[i][j]);
  }

  const int col = lane & 15, half = lane >> 4;
#pragma unroll
  for (int i = 0; i < 2; ++i)
#pragma unroll
    for (int j = 0; j < 4; ++j)
#pragma unroll
      for (int r = 0; r < 8; ++r) {
        int m = m0 + i * 16 + half * 8 + r;   // global row (b*SEQ + s)
        int n = n0 + j * 16 + col;            // global feature
        int bb = m >> 11, s = m & (SEQ - 1);
        int hh = n >> 6, d = n & (HDIM - 1);
        __bf16 v = (__bf16)acc[i][j][r];
        if (z == 0)
          Q[(((size_t)bb * NHEADS + hh) * SEQ + s) * HDIM + d] = v;
        else if (z == 1)
          K[(((size_t)bb * NHEADS + hh) * SEQ + s) * HDIM + d] = v;
        else
          Vt[(((size_t)bb * NHEADS + hh) * HDIM + d) * SEQ + s] = v;
      }
}

// ---------------------------------------------------------------------------
// 4) Causal flash attention.  One wave per (b, h, 32-query block): two
//    16-query tiles share every 32-key K/V chunk.  Because the query block
//    and the key chunk are both 32-aligned, BOTH tiles are active in EVERY
//    chunk -> no wasted WMMAs, no divergence.  Computes S^T = K * Q^T so
//    softmax stats are in-lane and the exp'd registers feed the P*V
//    A-fragment directly.  Per chunk: 8 fragment loads -> 16 WMMAs.
// ---------------------------------------------------------------------------
__global__ void __launch_bounds__(128, 1)
attention_kernel(const __bf16* __restrict__ Q,
                 const __bf16* __restrict__ K,
                 const __bf16* __restrict__ Vt,
                 __bf16* __restrict__ ctxb) {
  const int lane = threadIdx.x & 31;
  const int wave = threadIdx.x >> 5;
  const int wg = blockIdx.x * 4 + wave;         // 0..2047
  const int qblk = wg & (SEQ / 32 - 1);         // 32-query block in head
  const int h  = (wg >> 6) & (NHEADS - 1);
  const int b  = wg >> 10;
  const int q0 = qblk * 32;

  const __bf16* Qp = Q  + ((size_t)(b * NHEADS + h) * SEQ) * HDIM;   // [seq][64]
  const __bf16* Kp = K  + ((size_t)(b * NHEADS + h) * SEQ) * HDIM;   // [seq][64]
  const __bf16* Vp = Vt + ((size_t)(b * NHEADS + h) * HDIM) * SEQ;   // [64][seq]

  const int col = lane & 15, half = lane >> 4;
  const float scale = 0.125f;                   // 1/sqrt(64)

  // Q tiles as B fragments (d = 0..31 and 32..63)
  v16bf bq[2][2];
#pragma unroll
  for (int t = 0; t < 2; ++t) {
    bq[t][0] = load_b_frag(Qp, HDIM, q0 + t * 16, 0, lane);
    bq[t][1] = load_b_frag(Qp, HDIM, q0 + t * 16, 32, lane);
  }

  v8f ctx[2][4] = {};                           // 32q x 64d accumulator
  float m_run[2], l_run[2];
#pragma unroll
  for (int t = 0; t < 2; ++t) { m_run[t] = -__builtin_inff(); l_run[t] = 0.f; }

  for (int key0 = 0; key0 <= q0 + 31; key0 += 32) {
    __builtin_prefetch(Kp + (size_t)(key0 + 32) * HDIM, 0, 0);

    // Shared K chunk (two 16-key subtiles) for both query tiles
    v16bf ak00 = load_a_frag(Kp, HDIM, key0,      0,  lane);
    v16bf ak01 = load_a_frag(Kp, HDIM, key0,      32, lane);
    v16bf ak10 = load_a_frag(Kp, HDIM, key0 + 16, 0,  lane);
    v16bf ak11 = load_a_frag(Kp, HDIM, key0 + 16, 32, lane);

    // Phase 1: per-tile S^T, softmax, P fragment (keeps only ap[] live)
    v16bf ap[2];
#pragma unroll
    for (int t = 0; t < 2; ++t) {
      const int q_global = q0 + t * 16 + col;   // this lane's query column

      v8f zero = {};
      v8f s0 = wmma_bf16(ak01, bq[t][1], wmma_bf16(ak00, bq[t][0], zero));
      v8f s1 = wmma_bf16(ak11, bq[t][1], wmma_bf16(ak10, bq[t][0], zero));

      // scale + causal mask + chunk row-max (per query column)
      float p0[8], p1[8];
      float mx = -__builtin_inff();
#pragma unroll
      for (int r = 0; r < 8; ++r) {
        int k0r = key0 + half * 8 + r;
        int k1r = k0r + 16;
        float v0 = (k0r <= q_global) ? s0[r] * scale : -__builtin_inff();
        float v1 = (k1r <= q_global) ? s1[r] * scale : -__builtin_inff();
        p0[r] = v0; p1[r] = v1;
        mx = fmaxf(mx, fmaxf(v0, v1));
      }
      mx = fmaxf(mx, __shfl_xor(mx, 16));

      // online softmax update
      float m_new = fmaxf(m_run[t], mx);
      float alpha = __expf(m_run[t] - m_new);   // 0 on first chunk
      float rs = 0.f;
#pragma unroll
      for (int r = 0; r < 8; ++r) {
        p0[r] = __expf(p0[r] - m_new);
        p1[r] = __expf(p1[r] - m_new);
        rs += p0[r] + p1[r];
      }
      rs += __shfl_xor(rs, 16);
      l_run[t] = l_run[t] * alpha + rs;
      m_run[t] = m_new;

      // rescale ctx: ctx row M = half*8 + r, its alpha lives in lane M
#pragma unroll
      for (int r = 0; r < 8; ++r) {
        float ar = __shfl(alpha, half * 8 + r);
        ctx[t][0][r] *= ar; ctx[t][1][r] *= ar;
        ctx[t][2][r] *= ar; ctx[t][3][r] *= ar;
      }

      // Exp'd S^T registers are exactly the P A-fragment (16q x 32keys)
#pragma unroll
      for (int r = 0; r < 8; ++r) {
        ap[t][r] = (__bf16)p0[r]; ap[t][r + 8] = (__bf16)p1[r];
      }
    }

    // Phase 2: stream V fragments, each feeds both query tiles
#pragma unroll
    for (int dt = 0; dt < 4; ++dt) {
      v16bf bv = load_b_frag(Vp, SEQ, dt * 16, key0, lane);  // V^T rows = d
      ctx[0][dt] = wmma_bf16(ap[0], bv, ctx[0][dt]);
      ctx[1][dt] = wmma_bf16(ap[1], bv, ctx[1][dt]);
    }
  }

  // normalize by l and store ctx as [b][seq][h*64+d] bf16
#pragma unroll
  for (int t = 0; t < 2; ++t)
#pragma unroll
    for (int r = 0; r < 8; ++r) {
      float lr = __shfl(l_run[t], half * 8 + r);
      float inv = 1.f / lr;
      int qrow = q0 + t * 16 + half * 8 + r;
      size_t base = ((size_t)b * SEQ + qrow) * D_OUT + h * HDIM;
#pragma unroll
      for (int dt = 0; dt < 4; ++dt)
        ctxb[base + dt * 16 + col] = (__bf16)(ctx[t][dt][r] * inv);
    }
}

// ---------------------------------------------------------------------------
// 5) Output projection: out = ctx(4096x1024) * Wo^T + bo  (fp32 out),
//    32x64 tile per wave.
// ---------------------------------------------------------------------------
__global__ void __launch_bounds__(128, 1)
out_proj_kernel(const __bf16* __restrict__ ctxb,
                const __bf16* __restrict__ WoT,
                const float* __restrict__ bo,
                float* __restrict__ out) {
  const int lane = threadIdx.x & 31;
  const int wave = threadIdx.x >> 5;
  const int m0 = (blockIdx.x * 4 + wave) * 32;
  const int n0 = blockIdx.y * 64;

  v8f acc[2][4] = {};
  for (int k = 0; k < D_OUT; k += 32) {
    v16bf a[2], b[4];
#pragma unroll
    for (int i = 0; i < 2; ++i) a[i] = load_a_frag(ctxb, D_OUT, m0 + i * 16, k, lane);
#pragma unroll
    for (int j = 0; j < 4; ++j) b[j] = load_b_frag(WoT, D_OUT, n0 + j * 16, k, lane);
#pragma unroll
    for (int i = 0; i < 2; ++i)
#pragma unroll
      for (int j = 0; j < 4; ++j) acc[i][j] = wmma_bf16(a[i], b[j], acc[i][j]);
  }

  const int col = lane & 15, half = lane >> 4;
#pragma unroll
  for (int i = 0; i < 2; ++i)
#pragma unroll
    for (int j = 0; j < 4; ++j) {
      int n = n0 + j * 16 + col;
      float bias = bo[n];
#pragma unroll
      for (int r = 0; r < 8; ++r) {
        int m = m0 + i * 16 + half * 8 + r;
        out[(size_t)m * D_OUT + n] = acc[i][j][r] + bias;
      }
    }
}

// ---------------------------------------------------------------------------
extern "C" void kernel_launch(void* const* d_in, const int* in_sizes, int n_in,
                              void* d_out, int out_size, void* d_ws, size_t ws_size,
                              hipStream_t stream) {
  const float* x  = (const float*)d_in[0];
  const float* Wq = (const float*)d_in[1];
  const float* Wk = (const float*)d_in[2];
  const float* Wv = (const float*)d_in[3];
  const float* Wo = (const float*)d_in[4];
  const float* bo = (const float*)d_in[5];
  float* out = (float*)d_out;

  // workspace layout (bf16 buffers), ~48 MB total
  char* ws = (char*)d_ws;
  size_t off = 0;
  auto carve = [&](size_t bytes) {
    char* p = ws + off;
    off += (bytes + 255) & ~(size_t)255;
    return p;
  };
  const size_t ACT = (size_t)BN * D_IN * sizeof(__bf16);       // 8 MB
  const size_t WT  = (size_t)D_IN * D_OUT * sizeof(__bf16);    // 2 MB
  const size_t PH  = (size_t)BATCH * NHEADS * SEQ * HDIM * sizeof(__bf16); // 8 MB
  __bf16* xb   = (__bf16*)carve(ACT);
  __bf16* WqT  = (__bf16*)carve(WT);
  __bf16* WkT  = (__bf16*)carve(WT);
  __bf16* WvT  = (__bf16*)carve(WT);
  __bf16* WoT  = (__bf16*)carve(WT);
  __bf16* Qb   = (__bf16*)carve(PH);
  __bf16* Kb   = (__bf16*)carve(PH);
  __bf16* Vtb  = (__bf16*)carve(PH);
  __bf16* ctxb = (__bf16*)carve(ACT);

  const int n_x = BN * D_IN;
  cvt_x_kernel<<<(n_x + 255) / 256, 256, 0, stream>>>(x, xb, n_x);

  const int n_w = D_IN * D_OUT;
  transpose_w_kernel<<<n_w / 256, 256, 0, stream>>>(Wq, WqT);
  transpose_w_kernel<<<n_w / 256, 256, 0, stream>>>(Wk, WkT);
  transpose_w_kernel<<<n_w / 256, 256, 0, stream>>>(Wv, WvT);
  transpose_w_kernel<<<n_w / 256, 256, 0, stream>>>(Wo, WoT);

  // 32x64 tiles, 4 waves/block: block covers 128 rows x 64 cols
  qkv_proj_kernel<<<dim3(BN / 128, D_OUT / 64, 3), 128, 0, stream>>>(
      xb, WqT, WkT, WvT, Qb, Kb, Vtb);

  // one wave per 32-query block: 2*16*64 = 2048 waves, 4 per block
  attention_kernel<<<(BATCH * NHEADS * (SEQ / 32)) / 4, 128, 0, stream>>>(
      Qb, Kb, Vtb, ctxb);

  out_proj_kernel<<<dim3(BN / 128, D_OUT / 64), 128, 0, stream>>>(
      ctxb, WoT, bo, out);
}
